// RetICLBase_75840532512876
// MI455X (gfx1250) — matrix operand order repaired and checked by reading, not compiled, not run on hardware
//
#include <hip/hip_runtime.h>
#include <hip/hip_bf16.h>
#include <math.h>

// ---------------------------------------------------------------------------
// Problem constants (from reference): B=32, T=4, N=200000, EMB=768, H=800
// ---------------------------------------------------------------------------
#define BB   32
#define TT   4
#define NN   200000
#define EMB  768
#define HH   800
#define ROWS (BB * TT)   // 128
#define APAD 40          // LDS row stride (elements) to avoid bank conflicts

typedef __attribute__((ext_vector_type(16))) __bf16 v16bf;
typedef __attribute__((ext_vector_type(8)))  float  v8f;
typedef __attribute__((ext_vector_type(4)))  float  f4v;
typedef __attribute__((ext_vector_type(4)))  unsigned int u4v;

// ---------------------------------------------------------------------------
// Kernel 1: latent = tanh(cur@Wc + shifted@We + bh)  -> query(bf16), vals
// One block per batch b; all 4 timesteps processed together so Wc/We stream
// through the block exactly once (they stay L2-resident across blocks anyway).
// ---------------------------------------------------------------------------
__global__ void __launch_bounds__(256)
latent_query_vals_kernel(const float* __restrict__ cur,   // [B][EMB]
                         const float* __restrict__ ex,    // [B][T][EMB]
                         const float* __restrict__ Wc,    // [EMB][H]
                         const float* __restrict__ We,    // [EMB][H]
                         const float* __restrict__ bh,    // [H]
                         const float* __restrict__ bil,   // [H][EMB]
                         const float* __restrict__ vw,    // [H]
                         const float* __restrict__ vb,    // [1]
                         __bf16* __restrict__ qbf,        // [ROWS][EMB] bf16
                         float*  __restrict__ vals_out)   // [ROWS]
{
    const int b   = blockIdx.x;
    const int tid = threadIdx.x;

    __shared__ float curS[EMB];
    __shared__ float shS[TT][EMB];
    __shared__ float latS[TT][HH];
    __shared__ float red[256];

    for (int e = tid; e < EMB; e += 256) {
        curS[e]   = cur[b * EMB + e];
        shS[0][e] = 0.0f;                         // shifted[:,0] = 0
        shS[1][e] = ex[(b * TT + 0) * EMB + e];   // shifted[:,t] = ex[:,t-1]
        shS[2][e] = ex[(b * TT + 1) * EMB + e];
        shS[3][e] = ex[(b * TT + 2) * EMB + e];
    }
    __syncthreads();

    // latent accumulation: 4 h-values per thread (h = tid + 256*j), 4 t's.
    float acc[TT][4];
    #pragma unroll
    for (int t = 0; t < TT; ++t)
        #pragma unroll
        for (int j = 0; j < 4; ++j) {
            int h = tid + 256 * j;
            acc[t][j] = (h < HH) ? bh[h] : 0.0f;
        }

    for (int e = 0; e < EMB; ++e) {
        float c  = curS[e];
        float s0 = shS[0][e], s1 = shS[1][e], s2 = shS[2][e], s3 = shS[3][e];
        #pragma unroll
        for (int j = 0; j < 4; ++j) {
            int h = tid + 256 * j;
            if (h < HH) {
                float wc = Wc[e * HH + h];
                float we = We[e * HH + h];
                float base = c * wc;
                acc[0][j] += base + s0 * we;
                acc[1][j] += base + s1 * we;
                acc[2][j] += base + s2 * we;
                acc[3][j] += base + s3 * we;
            }
        }
    }

    #pragma unroll
    for (int t = 0; t < TT; ++t)
        #pragma unroll
        for (int j = 0; j < 4; ++j) {
            int h = tid + 256 * j;
            if (h < HH) latS[t][h] = tanhf(acc[t][j]);
        }
    __syncthreads();

    // value head: vals[b*T+t] = latent . vw + vb
    for (int t = 0; t < TT; ++t) {
        float p = 0.0f;
        for (int h = tid; h < HH; h += 256) p += latS[t][h] * vw[h];
        red[tid] = p;
        __syncthreads();
        for (int s = 128; s > 0; s >>= 1) {
            if (tid < s) red[tid] += red[tid + s];
            __syncthreads();
        }
        if (tid == 0) vals_out[b * TT + t] = red[0] + vb[0];
        __syncthreads();
    }

    // query = latent @ bilinear  -> bf16 (bilinear streamed once per block)
    for (int e = tid; e < EMB; e += 256) {
        float q0 = 0.f, q1 = 0.f, q2 = 0.f, q3 = 0.f;
        for (int h = 0; h < HH; ++h) {
            float w = bil[h * EMB + e];
            q0 += latS[0][h] * w;
            q1 += latS[1][h] * w;
            q2 += latS[2][h] * w;
            q3 += latS[3][h] * w;
        }
        qbf[(b * TT + 0) * EMB + e] = (__bf16)q0;
        qbf[(b * TT + 1) * EMB + e] = (__bf16)q1;
        qbf[(b * TT + 2) * EMB + e] = (__bf16)q2;
        qbf[(b * TT + 3) * EMB + e] = (__bf16)q3;
    }
}

// ---------------------------------------------------------------------------
// Kernel 2: acts[128, 200000] = query(bf16) @ all_ex^T (fp32->bf16 on the fly)
// 8 waves/block; wave owns 16 N-columns, covers ALL 128 M rows so all_ex is
// streamed from HBM exactly once (non-temporal).  The shared A K-chunk is
// staged in double-buffered LDS once per block (8x less L2 traffic than
// per-wave reloads) and fed to v_wmma_f32_16x16x32_bf16 via ds_load_b128.
// ---------------------------------------------------------------------------

// A fragment (16x32 bf16, M x K) from LDS: lane<16 holds K {0..15},
// lane>=16 holds K {16..31}; row M = m0 + (lane & 15).
__device__ inline v16bf load_a_frag_lds(const __bf16* __restrict__ abase,
                                        int m0, int lane)
{
    const int row = m0 + (lane & 15);
    const int sel = (lane < 16) ? 0 : 16;
    const __bf16* ptr = abase + row * APAD + sel;
    union { u4v u[2]; v16bf v; } a;
    a.u[0] = *reinterpret_cast<const u4v*>(ptr);        // K sel+0..7
    a.u[1] = *reinterpret_cast<const u4v*>(ptr + 8);    // K sel+8..15
    return a.v;
}

// B fragment (32x16 bf16, K x N): column N = lane & 15; lane<16 holds K
// k0..k0+15, lane>=16 holds K k0+16..k0+31 (16 contiguous elements of the
// all_ex row), streamed non-temporally and converted fp32 -> bf16 in regs.
__device__ inline v16bf load_b_frag(const float* __restrict__ all_ex,
                                    long long nrow, int k0, int lane)
{
    const int koff = k0 + ((lane < 16) ? 0 : 16);
    const f4v* p = reinterpret_cast<const f4v*>(all_ex + nrow * EMB + koff);
    f4v f[4];
    #pragma unroll
    for (int i = 0; i < 4; ++i) f[i] = __builtin_nontemporal_load(p + i);
    v16bf bfrag;
    #pragma unroll
    for (int i = 0; i < 4; ++i)
        #pragma unroll
        for (int j = 0; j < 4; ++j)
            bfrag[i * 4 + j] = (__bf16)f[i][j];
    return bfrag;
}

__global__ void __launch_bounds__(256)
gemm_acts_kernel(const __bf16* __restrict__ qbf,     // [128][768] bf16
                 const float*  __restrict__ all_ex,  // [N][768] fp32
                 const float*  __restrict__ bias,    // [1]
                 float*        __restrict__ out)     // acts [128][N]
{
    const int tid  = threadIdx.x;
    const int lane = tid & 31;
    const int wave = tid >> 5;
    const long long ncol_base = (long long)blockIdx.x * 128 + wave * 16;

    long long nrow = ncol_base + (lane & 15);
    if (nrow >= NN) nrow = NN - 1;   // uniform clamp: EXEC stays all-ones

    // Double-buffered A tile: [buf][row 0..127][32 K-elements, padded to 40]
    __shared__ alignas(16) __bf16 Atile[2][ROWS][APAD];

    // Each thread copies two 8-element (16 B) chunks per K-chunk:
    // chunk c in [0,512): row = c>>2, q = c&3  -> elements q*8..q*8+7
    const int c0 = tid * 2, c1 = c0 + 1;
    const int row0 = c0 >> 2, q0 = c0 & 3;
    const int row1 = c1 >> 2, q1 = c1 & 3;

    // stage K-chunk 0 into buffer 0
    *reinterpret_cast<u4v*>(&Atile[0][row0][q0 * 8]) =
        *reinterpret_cast<const u4v*>(qbf + row0 * EMB + q0 * 8);
    *reinterpret_cast<u4v*>(&Atile[0][row1][q1 * 8]) =
        *reinterpret_cast<const u4v*>(qbf + row1 * EMB + q1 * 8);
    __syncthreads();

    v8f acc[8];
    #pragma unroll
    for (int m = 0; m < 8; ++m)
        #pragma unroll
        for (int j = 0; j < 8; ++j) acc[m][j] = 0.0f;

    const int NK = EMB / 32;   // 24 K-chunks
    for (int kc = 0; kc < NK; ++kc) {
        const int p  = kc & 1;
        const int k0 = kc * 32;

        // fetch next A K-chunk into registers (overlaps with WMMA below)
        u4v nr0, nr1;
        const bool have_next = (kc + 1 < NK);
        if (have_next) {
            const int nk0 = k0 + 32;
            nr0 = *reinterpret_cast<const u4v*>(qbf + row0 * EMB + nk0 + q0 * 8);
            nr1 = *reinterpret_cast<const u4v*>(qbf + row1 * EMB + nk0 + q1 * 8);
        }

        // prefetch the streamed matrix a few chunks ahead
        if (k0 + 128 < EMB)
            __builtin_prefetch(all_ex + nrow * EMB + k0 + 128, 0, 3);

        v16bf bfrag = load_b_frag(all_ex, nrow, k0, lane);

        const __bf16* abase = &Atile[p][0][0];
        #pragma unroll
        for (int m = 0; m < 8; ++m) {
            v16bf afrag = load_a_frag_lds(abase, m * 16, lane);
            acc[m] = __builtin_amdgcn_wmma_f32_16x16x32_bf16(
                         /*neg_a=*/false, afrag,
                         /*neg_b=*/false, bfrag,
                         /*c_mod=*/(short)0, acc[m],
                         /*reuse_a=*/false, /*reuse_b=*/false);
        }

        // commit next chunk to the other buffer; barrier orders both the
        // stores (visible next iter) and this iter's LDS reads (safe reuse)
        if (have_next) {
            *reinterpret_cast<u4v*>(&Atile[p ^ 1][row0][q0 * 8]) = nr0;
            *reinterpret_cast<u4v*>(&Atile[p ^ 1][row1][q1 * 8]) = nr1;
        }
        __syncthreads();
    }

    const float bia = bias[0];
    const long long col = ncol_base + (lane & 15);
    if (col < NN) {
        const int mrow_off = (lane < 16) ? 0 : 8;   // C/D layout
        #pragma unroll
        for (int m = 0; m < 8; ++m)
            #pragma unroll
            for (int j = 0; j < 8; ++j) {
                long long row = m * 16 + j + mrow_off;
                __builtin_nontemporal_store(acc[m][j] + bia,
                                            &out[row * (long long)NN + col]);
            }
    }
}

// ---------------------------------------------------------------------------
// Kernel 3: scatter -inf masks.  pairs (t,u) for T=4:
// (1,0),(2,0),(3,0),(2,1),(3,1),(3,2); acts[b, t, policy_idx[b,u]] = -inf
// ---------------------------------------------------------------------------
__global__ void mask_kernel(const int* __restrict__ pidx,  // [B][T]
                            float* __restrict__ out)       // acts [128][N]
{
    int i = blockIdx.x * blockDim.x + threadIdx.x;
    if (i >= BB * 6) return;
    const int tA[6] = {1, 2, 3, 2, 3, 3};
    const int uA[6] = {0, 0, 0, 1, 1, 2};
    int b = i / 6, p = i % 6;
    int col = pidx[b * TT + uA[p]];
    long long row = b * TT + tA[p];
    if (col >= 0 && col < NN)
        out[row * (long long)NN + col] = -__builtin_huge_valf();
}

// ---------------------------------------------------------------------------
// Launch
// ---------------------------------------------------------------------------
extern "C" void kernel_launch(void* const* d_in, const int* in_sizes, int n_in,
                              void* d_out, int out_size, void* d_ws, size_t ws_size,
                              hipStream_t stream)
{
    const float* cur    = (const float*)d_in[0];   // [B][EMB]
    const float* ex     = (const float*)d_in[1];   // [B][T][EMB]
    const int*   pidx   = (const int*)  d_in[2];   // [B][T]
    const float* all_ex = (const float*)d_in[3];   // [N][EMB]
    const float* bil    = (const float*)d_in[4];   // [H][EMB]
    const float* bias   = (const float*)d_in[5];   // [1]
    const float* Wc     = (const float*)d_in[6];   // [EMB][H]
    const float* We     = (const float*)d_in[7];   // [EMB][H]
    const float* bh     = (const float*)d_in[8];   // [H]
    const float* vw     = (const float*)d_in[9];   // [H]
    const float* vb     = (const float*)d_in[10];  // [1]

    float*  acts = (float*)d_out;                                  // [128][N]
    float*  vals = (float*)d_out + (long long)ROWS * NN;           // [128]
    __bf16* qbf  = (__bf16*)d_ws;                                  // [128][768]

    latent_query_vals_kernel<<<BB, 256, 0, stream>>>(
        cur, ex, Wc, We, bh, bil, vw, vb, qbf, vals);

    const int ngrids = (NN + 127) / 128;   // 1563
    gemm_acts_kernel<<<ngrids, 256, 0, stream>>>(qbf, all_ex, bias, acts);

    mask_kernel<<<1, 256, 0, stream>>>(pidx, acts);
}
